// EdgeGCN_K_Set2Set_5076651344426
// MI455X (gfx1250) — matrix operand order repaired and checked by reading, not compiled, not run on hardware
//
#include <hip/hip_runtime.h>
#include <hip/hip_bf16.h>

typedef __attribute__((ext_vector_type(16))) _Float16 v16h;
typedef __attribute__((ext_vector_type(8)))  _Float16 v8h;
typedef __attribute__((ext_vector_type(8)))  float    v8f;

#define NNODES 65536
#define NEDGES 1048576
#define NF 32
#define EFD 16
#define HD 64
#define BG 256
#define S2S_STEPS 12

static __device__ __forceinline__ v8f wmma_f16(v16h a, v16h b, v8f c) {
  // D(16x16,f32) = A(16x32,f16) * B(32x16,f16) + C
  return __builtin_amdgcn_wmma_f32_16x16x32_f16(
      /*neg_a=*/false, a, /*neg_b=*/false, b,
      /*c_mod=*/(short)0, c, /*reuse_a=*/false, /*reuse_b=*/false);
}

static __device__ __forceinline__ v16h cat8(v8h lo, v8h hi) {
  return __builtin_shufflevector(lo, hi, 0, 1, 2, 3, 4, 5, 6, 7,
                                 8, 9, 10, 11, 12, 13, 14, 15);
}

// ---------------------------------------------------------------------------
// sup = (relu?)(x) @ W      x:[N,KDIM] f32, W:[KDIM,64] f32, sup:[N,64] f32
// one wave -> 16-row tile; W transposed into LDS as f16 so B-fragments are
// contiguous; x rows loaded as float4; output bounced via LDS -> float4 stores
// ---------------------------------------------------------------------------
template <int KDIM, bool RELU_IN>
__global__ void node_transform_kernel(const float* __restrict__ x,
                                      const float* __restrict__ W,
                                      float* __restrict__ sup) {
  __shared__ __align__(32) _Float16 sWt[64 * KDIM];  // [col][k]
  __shared__ __align__(16) float ldsOut[8][16 * 64];
  const int tid = threadIdx.x;
  for (int i = tid; i < KDIM * 64; i += blockDim.x) {
    const int col = i & 63, k = i >> 6;  // coalesced read of W[k][col]
    sWt[col * KDIM + k] = (_Float16)W[k * 64 + col];
  }
  __syncthreads();

  const int wave = tid >> 5, lane = tid & 31;
  const int rowbase = (blockIdx.x * (blockDim.x >> 5) + wave) * 16;
  const int M = lane & 15;
  const int hi = (lane >> 4);      // 0: lanes 0-15, 1: lanes 16-31
  const int kbA = hi * 8;          // A-frag K base within 16-chunk
  const int kbB = hi * 16;         // unused; B-frags use kbA like A
  const int col16 = lane & 15;

  v8f acc[4];
#pragma unroll
  for (int t = 0; t < 4; ++t)
#pragma unroll
    for (int r = 0; r < 8; ++r) acc[t][r] = 0.f;

  const float4* xr4 = (const float4*)(x + (size_t)(rowbase + M) * KDIM);
#pragma unroll
  for (int kc = 0; kc < KDIM; kc += 32) {
    float4 f0 = xr4[(kc + kbA) >> 2];
    float4 f1 = xr4[((kc + kbA) >> 2) + 1];
    float4 f2 = xr4[(kc + 16 + kbA) >> 2];
    float4 f3 = xr4[((kc + 16 + kbA) >> 2) + 1];
    if (RELU_IN) {
      f0.x = fmaxf(f0.x, 0.f); f0.y = fmaxf(f0.y, 0.f); f0.z = fmaxf(f0.z, 0.f); f0.w = fmaxf(f0.w, 0.f);
      f1.x = fmaxf(f1.x, 0.f); f1.y = fmaxf(f1.y, 0.f); f1.z = fmaxf(f1.z, 0.f); f1.w = fmaxf(f1.w, 0.f);
      f2.x = fmaxf(f2.x, 0.f); f2.y = fmaxf(f2.y, 0.f); f2.z = fmaxf(f2.z, 0.f); f2.w = fmaxf(f2.w, 0.f);
      f3.x = fmaxf(f3.x, 0.f); f3.y = fmaxf(f3.y, 0.f); f3.z = fmaxf(f3.z, 0.f); f3.w = fmaxf(f3.w, 0.f);
    }
    v16h a;
    a[0] = (_Float16)f0.x; a[1] = (_Float16)f0.y; a[2] = (_Float16)f0.z; a[3] = (_Float16)f0.w;
    a[4] = (_Float16)f1.x; a[5] = (_Float16)f1.y; a[6] = (_Float16)f1.z; a[7] = (_Float16)f1.w;
    a[8] = (_Float16)f2.x; a[9] = (_Float16)f2.y; a[10] = (_Float16)f2.z; a[11] = (_Float16)f2.w;
    a[12] = (_Float16)f3.x; a[13] = (_Float16)f3.y; a[14] = (_Float16)f3.z; a[15] = (_Float16)f3.w;
#pragma unroll
    for (int t = 0; t < 4; ++t) {
      const int col = t * 16 + col16;
      const v16h b = cat8(*(const v8h*)(sWt + col * KDIM + kc + kbA), *(const v8h*)(sWt + col * KDIM + kc + 16 + kbA));
      acc[t] = wmma_f16(a, b, acc[t]);
    }
  }
  // coalesce stores through LDS
  float* myO = ldsOut[wave];
#pragma unroll
  for (int t = 0; t < 4; ++t)
#pragma unroll
    for (int r = 0; r < 8; ++r)
      myO[(r + hi * 8) * 64 + t * 16 + col16] = acc[t][r];
  __syncthreads();
  for (int i = lane; i < 256; i += 32) {
    const int row = i >> 4, c4 = i & 15;
    ((float4*)(sup + (size_t)(rowbase + row) * 64))[c4] = ((const float4*)myO)[i];
  }
}

// ---------------------------------------------------------------------------
__global__ void init_bias_kernel(float* __restrict__ y,
                                 const float* __restrict__ b, int n) {
  const int i = blockIdx.x * blockDim.x + threadIdx.x;
  if (i < n) y[i] = b[i & 63];
}

// ---------------------------------------------------------------------------
// Fused:  enc = relu(ef@W1+b1)@W2+b2  (regs/LDS only, never hits HBM)
//         msg = sup[Esrc] * enc ; atomic scatter-add into y[Etgt]
// 128 threads = 4 waves; one 16-edge tile per wave.
// Weights transposed (and W1 zero-padded to K=32) in LDS: all fragment loads
// are contiguous vector LDS loads; no divergent predication anywhere.
// ---------------------------------------------------------------------------
__global__ void edge_layer_kernel(const float* __restrict__ ef,
                                  const float* __restrict__ W1,
                                  const float* __restrict__ b1,
                                  const float* __restrict__ W2,
                                  const float* __restrict__ b2,
                                  const float* __restrict__ sup,
                                  const int* __restrict__ Esrc,
                                  const int* __restrict__ Etgt,
                                  float* __restrict__ y) {
  __shared__ __align__(32) _Float16 sW1t[64 * 32];  // [col][k], k>=16 zero
  __shared__ __align__(32) _Float16 sW2t[64 * 64];  // [col][k]
  __shared__ float sB1[64], sB2[64];
  __shared__ __align__(16) _Float16 ldsH[4][16 * 64];  // hidden (f16, row-major)
  __shared__ __align__(16) float    ldsF[4][16 * 64];  // out tile (f32)

  const int tid = threadIdx.x;
  for (int i = tid; i < 64 * 16; i += 128) {
    const int col = i & 63, k = i >> 6;              // k 0..15
    sW1t[col * 32 + k] = (_Float16)W1[k * 64 + col]; // coalesced global read
    sW1t[col * 32 + 16 + k] = (_Float16)0.f;         // zero pad K=16..31
  }
  for (int i = tid; i < 64 * 64; i += 128) {
    const int col = i & 63, k = i >> 6;
    sW2t[col * 64 + k] = (_Float16)W2[k * 64 + col];
  }
  if (tid < 64) { sB1[tid] = b1[tid]; sB2[tid] = b2[tid]; }

  const int wave = tid >> 5, lane = tid & 31;
  const int ebase = (blockIdx.x * 4 + wave) * 16;
  const int M = lane & 15;
  const int hi = (lane >> 4);
  const int kbA = hi * 8;
  const int kbB = hi * 16;
  const int col16 = lane & 15;
  _Float16* myH = ldsH[wave];
  float* myF = ldsF[wave];

  // warm L2/L0 for the gather rows while the MLP runs
  if (lane < 16) {
    const int e = ebase + lane;
    __builtin_prefetch(sup + (size_t)Esrc[e] * 64, 0, 3);
  }
  __syncthreads();

  // ---- GEMM1: hidden = relu(ef @ W1 + b1), K=16 zero-padded to 32 ----
  v16h a1;
  {
    const float4* efr4 = (const float4*)(ef + (size_t)(ebase + M) * EFD);
    const float4 f0 = efr4[kbA >> 2];
    const float4 f1 = efr4[(kbA >> 2) + 1];
    a1[0] = (_Float16)f0.x; a1[1] = (_Float16)f0.y; a1[2] = (_Float16)f0.z; a1[3] = (_Float16)f0.w;
    a1[4] = (_Float16)f1.x; a1[5] = (_Float16)f1.y; a1[6] = (_Float16)f1.z; a1[7] = (_Float16)f1.w;
#pragma unroll
    for (int j = 8; j < 16; ++j) a1[j] = (_Float16)0.f;  // K=16..31 pad
  }
#pragma unroll
  for (int t = 0; t < 4; ++t) {
    const int col = t * 16 + col16;
    const v16h b = cat8(*(const v8h*)(sW1t + col * 32 + kbA), *(const v8h*)(sW1t + col * 32 + 16 + kbA));
    v8f c;
    const float bias = sB1[col];
#pragma unroll
    for (int r = 0; r < 8; ++r) c[r] = bias;
    c = wmma_f16(a1, b, c);
#pragma unroll
    for (int r = 0; r < 8; ++r) {
      const float v = fmaxf(c[r], 0.f);
      myH[(r + hi * 8) * 64 + col] = (_Float16)v;  // row-major 16x64
    }
  }
  __syncthreads();

  // ---- GEMM2: out = hidden @ W2 + b2, K=64 ----
  v16h a2[2];
#pragma unroll
  for (int kc2 = 0; kc2 < 2; ++kc2) {
    const _Float16* hrow = myH + M * 64 + kc2 * 32;
    const v8h lo = *(const v8h*)(hrow + kbA);
    const v8h hi8 = *(const v8h*)(hrow + 16 + kbA);
    a2[kc2] = cat8(lo, hi8);
  }
  v8f o[4];
#pragma unroll
  for (int t = 0; t < 4; ++t) {
    const int col = t * 16 + col16;
    const float bias = sB2[col];
#pragma unroll
    for (int r = 0; r < 8; ++r) o[t][r] = bias;
#pragma unroll
    for (int kc2 = 0; kc2 < 2; ++kc2) {
      const v16h b = cat8(*(const v8h*)(sW2t + col * 64 + kc2 * 32 + kbA), *(const v8h*)(sW2t + col * 64 + kc2 * 32 + 16 + kbA));
      o[t] = wmma_f16(a2[kc2], b, o[t]);
    }
  }

  // ---- stage out tile, then gather-modulate-scatter ----
#pragma unroll
  for (int t = 0; t < 4; ++t) {
    const int col = t * 16 + col16;
#pragma unroll
    for (int r = 0; r < 8; ++r) myF[(r + hi * 8) * 64 + col] = o[t][r];
  }
  __syncthreads();
  for (int i = lane; i < 16 * 64; i += 32) {
    const int row = i >> 6, col = i & 63;
    const int e = ebase + row;
    const int s = Esrc[e], tg = Etgt[e];
    const float msg = myF[i] * sup[(size_t)s * 64 + col];
    atomicAdd(y + (size_t)tg * 64 + col, msg);
  }
}

// ---------------------------------------------------------------------------
// Set2Set pieces
// ---------------------------------------------------------------------------
__global__ void zero_kernel(float* __restrict__ p, int n) {
  const int i = blockIdx.x * blockDim.x + threadIdx.x;
  if (i < n) p[i] = 0.f;
}

__global__ void lstm_kernel(const float* __restrict__ Wih,  // [256,128]
                            const float* __restrict__ Whh,  // [256,64]
                            const float* __restrict__ bih,
                            const float* __restrict__ bhh,
                            const float* __restrict__ r,
                            float* __restrict__ h, float* __restrict__ cst) {
  __shared__ float q[64], rr[64];
  const int b = blockIdx.x, j = threadIdx.x;  // 64 threads
  q[j] = h[b * 64 + j];
  rr[j] = r[b * 64 + j];
  __syncthreads();
  float g[4];
#pragma unroll
  for (int gi = 0; gi < 4; ++gi) {
    const int row = gi * 64 + j;
    float s = bih[row] + bhh[row];
    const float* wih = Wih + row * 128;
    const float* whh = Whh + row * 64;
    for (int k = 0; k < 64; ++k)
      s += wih[k] * q[k] + wih[64 + k] * rr[k] + whh[k] * q[k];
    g[gi] = s;
  }
  const float ig = 1.f / (1.f + __expf(-g[0]));
  const float fg = 1.f / (1.f + __expf(-g[1]));
  const float gg = tanhf(g[2]);
  const float og = 1.f / (1.f + __expf(-g[3]));
  const float cn = fg * cst[b * 64 + j] + ig * gg;
  cst[b * 64 + j] = cn;
  h[b * 64 + j] = og * tanhf(cn);
}

static __device__ __forceinline__ unsigned f2key(float x) {
  const unsigned u = __float_as_uint(x);
  return (u & 0x80000000u) ? ~u : (u | 0x80000000u);
}
static __device__ __forceinline__ float key2f(unsigned k) {
  return __uint_as_float((k & 0x80000000u) ? (k ^ 0x80000000u) : ~k);
}

__global__ void attn_reset_kernel(unsigned* __restrict__ emax,
                                  float* __restrict__ denom,
                                  float* __restrict__ r) {
  const int i = blockIdx.x * blockDim.x + threadIdx.x;
  if (i < BG * HD) r[i] = 0.f;
  if (i < BG) { emax[i] = 0u; denom[i] = 0.f; }
}

__global__ void attn_e_kernel(const float* __restrict__ x,
                              const float* __restrict__ h,
                              const int* __restrict__ batch,
                              float* __restrict__ eArr,
                              unsigned* __restrict__ emax) {
  const int wave = threadIdx.x >> 5, lane = threadIdx.x & 31;
  const int n = blockIdx.x * 8 + wave;
  const int b = batch[n];
  const float* xr = x + (size_t)n * 64;
  const float* hr = h + b * 64;
  float s = xr[lane] * hr[lane] + xr[lane + 32] * hr[lane + 32];
#pragma unroll
  for (int off = 16; off; off >>= 1) s += __shfl_xor(s, off, 32);
  if (lane == 0) {
    eArr[n] = s;
    atomicMax(&emax[b], f2key(s));
  }
}

__global__ void attn_denom_kernel(const int* __restrict__ batch,
                                  const unsigned* __restrict__ emax,
                                  float* __restrict__ eArr,
                                  float* __restrict__ denom) {
  const int n = blockIdx.x * blockDim.x + threadIdx.x;
  if (n >= NNODES) return;
  const int b = batch[n];
  const float ex = __expf(eArr[n] - key2f(emax[b]));
  eArr[n] = ex;
  atomicAdd(&denom[b], ex);
}

__global__ void attn_r_kernel(const float* __restrict__ x,
                              const int* __restrict__ batch,
                              const float* __restrict__ eArr,
                              const float* __restrict__ denom,
                              float* __restrict__ r) {
  const int wave = threadIdx.x >> 5, lane = threadIdx.x & 31;
  const int n = blockIdx.x * 8 + wave;
  const int b = batch[n];
  const float a = eArr[n] / denom[b];
  const float* xr = x + (size_t)n * 64;
  atomicAdd(&r[b * 64 + lane], a * xr[lane]);
  atomicAdd(&r[b * 64 + lane + 32], a * xr[lane + 32]);
}

__global__ void copy_kernel(const float* __restrict__ src,
                            float* __restrict__ dst, int n) {
  const int i = blockIdx.x * blockDim.x + threadIdx.x;
  if (i < n) dst[i] = src[i];
}

// ---------------------------------------------------------------------------
extern "C" void kernel_launch(void* const* d_in, const int* in_sizes, int n_in,
                              void* d_out, int out_size, void* d_ws,
                              size_t ws_size, hipStream_t stream) {
  const float* nodef = (const float*)d_in[0];
  const float* edgef = (const float*)d_in[1];
  const int* Esrc = (const int*)d_in[2];
  const int* Etgt = (const int*)d_in[3];
  const int* batch = (const int*)d_in[4];
  const float* eeW1[3] = {(const float*)d_in[5], (const float*)d_in[9],  (const float*)d_in[13]};
  const float* eeB1[3] = {(const float*)d_in[6], (const float*)d_in[10], (const float*)d_in[14]};
  const float* eeW2[3] = {(const float*)d_in[7], (const float*)d_in[11], (const float*)d_in[15]};
  const float* eeB2[3] = {(const float*)d_in[8], (const float*)d_in[12], (const float*)d_in[16]};
  const float* gcW[3] = {(const float*)d_in[17], (const float*)d_in[19], (const float*)d_in[21]};
  const float* gcB[3] = {(const float*)d_in[18], (const float*)d_in[20], (const float*)d_in[22]};
  const float* lstm_Wih = (const float*)d_in[23];
  const float* lstm_Whh = (const float*)d_in[24];
  const float* lstm_bih = (const float*)d_in[25];
  const float* lstm_bhh = (const float*)d_in[26];

  float* sup  = (float*)d_ws;                 // N*64
  float* bufA = sup + (size_t)NNODES * 64;    // N*64
  float* bufB = bufA + (size_t)NNODES * 64;   // N*64
  float* h = bufB + (size_t)NNODES * 64;      // B*64
  float* c = h + BG * 64;
  float* r = c + BG * 64;
  float* eArr = r + BG * 64;                  // N
  unsigned* emax = (unsigned*)(eArr + NNODES);
  float* denom = (float*)(emax + BG);

  const int NT_BLK = 256;              // 8 waves -> 128 rows/block
  const int nt_grid = NNODES / 128;
  const int edge_grid = NEDGES / 64;   // 4 waves * 16 edges
  const int ny = NNODES * 64;
  const int init_grid = (ny + 255) / 256;

  node_transform_kernel<NF, false><<<nt_grid, NT_BLK, 0, stream>>>(nodef, gcW[0], sup);
  init_bias_kernel<<<init_grid, 256, 0, stream>>>(bufA, gcB[0], ny);
  edge_layer_kernel<<<edge_grid, 128, 0, stream>>>(edgef, eeW1[0], eeB1[0], eeW2[0], eeB2[0],
                                                   sup, Esrc, Etgt, bufA);
  node_transform_kernel<HD, true><<<nt_grid, NT_BLK, 0, stream>>>(bufA, gcW[1], sup);
  init_bias_kernel<<<init_grid, 256, 0, stream>>>(bufB, gcB[1], ny);
  edge_layer_kernel<<<edge_grid, 128, 0, stream>>>(edgef, eeW1[1], eeB1[1], eeW2[1], eeB2[1],
                                                   sup, Esrc, Etgt, bufB);
  node_transform_kernel<HD, true><<<nt_grid, NT_BLK, 0, stream>>>(bufB, gcW[2], sup);
  init_bias_kernel<<<init_grid, 256, 0, stream>>>(bufA, gcB[2], ny);
  edge_layer_kernel<<<edge_grid, 128, 0, stream>>>(edgef, eeW1[2], eeB1[2], eeW2[2], eeB2[2],
                                                   sup, Esrc, Etgt, bufA);
  const float* xfin = bufA;

  zero_kernel<<<(3 * BG * 64 + 255) / 256, 256, 0, stream>>>(h, 3 * BG * 64);
  for (int step = 0; step < S2S_STEPS; ++step) {
    lstm_kernel<<<BG, 64, 0, stream>>>(lstm_Wih, lstm_Whh, lstm_bih, lstm_bhh, r, h, c);
    if (step < S2S_STEPS - 1) {  // last step's attention doesn't affect output
      attn_reset_kernel<<<(BG * 64 + 255) / 256, 256, 0, stream>>>(emax, denom, r);
      attn_e_kernel<<<NNODES / 8, 256, 0, stream>>>(xfin, h, batch, eArr, emax);
      attn_denom_kernel<<<(NNODES + 255) / 256, 256, 0, stream>>>(batch, emax, eArr, denom);
      attn_r_kernel<<<NNODES / 8, 256, 0, stream>>>(xfin, batch, eArr, denom, r);
    }
  }
  copy_kernel<<<(BG * 64 + 255) / 256, 256, 0, stream>>>(h, (float*)d_out, BG * 64);
}
